// TransformerBlock_5351529251405
// MI455X (gfx1250) — matrix operand order repaired and checked by reading, not compiled, not run on hardware
//
#include <hip/hip_runtime.h>
#include <hip/hip_bf16.h>

// ---------------------------------------------------------------------------
// Transformer block for MI455X (gfx1250): bf16 WMMA everywhere, flash attn
// with transposed-score softmax (in-lane reductions), async-LDS GEMM staging.
// B=2, S=2048, H=1024, NH=16, HD=64, F=4096.
// ---------------------------------------------------------------------------

typedef __bf16 bf16;
typedef __bf16 v16bf __attribute__((ext_vector_type(16)));
typedef __bf16 v8bf  __attribute__((ext_vector_type(8)));
typedef float  v8f   __attribute__((ext_vector_type(8)));

#define WMMA_BF16(a, b, c) \
  __builtin_amdgcn_wmma_f32_16x16x32_bf16(false, (a), false, (b), (short)0, (c), false, false)

static constexpr int Bc = 2;
static constexpr int Sc = 2048;
static constexpr int Hc = 1024;
static constexpr int NHc = 16;
static constexpr int HDc = 64;
static constexpr int Fc = 4096;
static constexpr int Mrows = Bc * Sc;     // 4096 token rows

// ---------------------------------------------------------------------------
// LayerNorm: fp32 row [1024] -> bf16 row. 256 threads/row, 4 elems/thread.
// ---------------------------------------------------------------------------
__global__ void ln_kernel(const float* __restrict__ x, const float* __restrict__ g,
                          const float* __restrict__ b, bf16* __restrict__ out) {
  const int row = blockIdx.x;
  const float4 v = ((const float4*)(x + (size_t)row * Hc))[threadIdx.x];
  float s  = v.x + v.y + v.z + v.w;
  float ss = v.x * v.x + v.y * v.y + v.z * v.z + v.w * v.w;
#pragma unroll
  for (int m = 1; m < 32; m <<= 1) {
    s  += __shfl_xor(s,  m, 32);
    ss += __shfl_xor(ss, m, 32);
  }
  __shared__ float rs[8], rss[8];
  if ((threadIdx.x & 31) == 0) { rs[threadIdx.x >> 5] = s; rss[threadIdx.x >> 5] = ss; }
  __syncthreads();
  s = 0.f; ss = 0.f;
#pragma unroll
  for (int i = 0; i < 8; i++) { s += rs[i]; ss += rss[i]; }
  const float mu   = s * (1.f / Hc);
  const float var  = ss * (1.f / Hc) - mu * mu;
  const float rstd = rsqrtf(var + 1e-5f);
  const float4 gv = ((const float4*)g)[threadIdx.x];
  const float4 bv = ((const float4*)b)[threadIdx.x];
  bf16* o = out + (size_t)row * Hc + threadIdx.x * 4;
  o[0] = (bf16)((v.x - mu) * rstd * gv.x + bv.x);
  o[1] = (bf16)((v.y - mu) * rstd * gv.y + bv.y);
  o[2] = (bf16)((v.z - mu) * rstd * gv.z + bv.z);
  o[3] = (bf16)((v.w - mu) * rstd * gv.w + bv.w);
}

// ---------------------------------------------------------------------------
// Weight transpose + fp32->bf16: W[K,N] -> Wt[N,K]. 32x32 tiles via LDS.
// ---------------------------------------------------------------------------
__global__ void transpose_convert(const float* __restrict__ W, bf16* __restrict__ Wt,
                                  int K, int N) {
  __shared__ float tile[32][33];
  const int n0 = blockIdx.x * 32, k0 = blockIdx.y * 32;
  const int tx = threadIdx.x & 31, ty = threadIdx.x >> 5;   // 256 thr: ty 0..7
#pragma unroll
  for (int i = 0; i < 4; i++)
    tile[ty + 8 * i][tx] = W[(size_t)(k0 + ty + 8 * i) * N + n0 + tx];
  __syncthreads();
#pragma unroll
  for (int i = 0; i < 4; i++)
    Wt[(size_t)(n0 + ty + 8 * i) * K + k0 + tx] = (bf16)tile[tx][ty + 8 * i];
}

// ---------------------------------------------------------------------------
// WMMA GEMM: C[M,N] = A[M,K](bf16) @ Wt[N,K](bf16) + bias, fused epilogue.
// Block: 128 thr (4 waves). BM=64, BN=128 (32 cols/wave), BK=32.
// A tile staged to LDS with CDNA5 async-LDS loads (dbl-buffered).
// ---------------------------------------------------------------------------
static constexpr int MODE_HEAD   = 0;  // bf16 out, [B,NH,S,HD] layout (Q, K)
static constexpr int MODE_VT     = 1;  // bf16 out, [B,NH,HD,S] layout (V^T)
static constexpr int MODE_RESF32 = 2;  // f32 out, + residual (out-proj, final)
static constexpr int MODE_GELU   = 3;  // bf16 out, exact GELU (FFN1)

template <int MODE>
__global__ void gemm_wmma(const bf16* __restrict__ A, const bf16* __restrict__ Wt,
                          const float* __restrict__ bias, const float* __restrict__ res,
                          void* __restrict__ outp, int M, int N, int K) {
  __shared__ alignas(32) bf16 asmem[2][64 * 32];
  const int lane = threadIdx.x & 31;
  const int wave = threadIdx.x >> 5;
  const int m0   = blockIdx.x * 64;
  const int n0   = blockIdx.y * 128 + wave * 32;
  const int lm   = lane & 15;
  const int kb8  = (lane >> 4) * 8;    // unused here (A-frag loads use kb16)
  const int kb16 = (lane >> 4) * 16;   // A/B-frag K base (contiguous k = 16h + s)
  const int ar   = threadIdx.x >> 1;   // LDS stage: row 0..63
  const int ac   = (threadIdx.x & 1) * 16;

  v8f acc[2][4] = {};
  const bf16* bptr0 = Wt + (size_t)(n0 + lm) * K + kb16;
  const bf16* bptr1 = Wt + (size_t)(n0 + 16 + lm) * K + kb16;

  for (int k0 = 0; k0 < K; k0 += 32) {
    const int buf = (k0 >> 5) & 1;
    // CDNA5 async global->LDS copy of the A tile (tracked by ASYNCcnt)
    {
      const unsigned int ldsa = (unsigned int)(uintptr_t)&asmem[buf][ar * 32 + ac];
      const unsigned long long ga =
          (unsigned long long)(uintptr_t)(A + (size_t)(m0 + ar) * K + k0 + ac);
      asm volatile("global_load_async_to_lds_b128 %0, %1, off"
                   :: "v"(ldsa), "v"(ga) : "memory");
      asm volatile("s_wait_asynccnt 0" ::: "memory");
    }
    __syncthreads();
    // prefetch next B rows while this step computes
    if (k0 + 32 < K) {
      __builtin_prefetch(bptr0 + k0 + 32, 0, 1);
      __builtin_prefetch(bptr1 + k0 + 32, 0, 1);
    }
    const v16bf bfrag0 = *(const v16bf*)(bptr0 + k0);
    const v16bf bfrag1 = *(const v16bf*)(bptr1 + k0);
#pragma unroll
    for (int mt = 0; mt < 4; mt++) {
      union { v16bf v; v8bf h[2]; } af;
      af.h[0] = *(const v8bf*)&asmem[buf][(mt * 16 + lm) * 32 + kb16];
      af.h[1] = *(const v8bf*)&asmem[buf][(mt * 16 + lm) * 32 + kb16 + 8];
      acc[0][mt] = WMMA_BF16(af.v, bfrag0, acc[0][mt]);
      acc[1][mt] = WMMA_BF16(af.v, bfrag1, acc[1][mt]);
    }
  }

#pragma unroll
  for (int nt = 0; nt < 2; nt++) {
    const int col = n0 + nt * 16 + lm;
    const float bv = bias[col];
#pragma unroll
    for (int mt = 0; mt < 4; mt++) {
#pragma unroll
      for (int r = 0; r < 8; r++) {
        const int row = m0 + mt * 16 + r + ((lane >> 4) << 3);  // D-layout M map
        const float val = acc[nt][mt][r] + bv;
        if (MODE == MODE_RESF32) {
          float* o = (float*)outp;
          o[(size_t)row * N + col] = val + res[(size_t)row * N + col];
        } else if (MODE == MODE_GELU) {
          bf16* o = (bf16*)outp;
          o[(size_t)row * N + col] = (bf16)(0.5f * val * (1.f + erff(val * 0.70710678118f)));
        } else if (MODE == MODE_HEAD) {
          bf16* o = (bf16*)outp;
          const int bb = row >> 11, s = row & (Sc - 1), nh = col >> 6, hd = col & 63;
          o[(((size_t)(bb * NHc + nh) * Sc + s) << 6) + hd] = (bf16)val;
        } else {  // MODE_VT
          bf16* o = (bf16*)outp;
          const int bb = row >> 11, s = row & (Sc - 1), nh = col >> 6, hd = col & 63;
          o[((size_t)(bb * NHc + nh) * HDc + hd) * Sc + s] = (bf16)val;
        }
      }
    }
  }
}

// ---------------------------------------------------------------------------
// Flash attention, transposed scores: S^T = K_chunk @ Q^T so each lane owns
// one query row's scores -> in-lane softmax reductions (no bpermute storms).
// q,k: [B,NH,S,64] bf16.  vt: [B,NH,64,S] bf16.  ao: [B,S,H] bf16 (merged).
// Block: 128 thr (4 independent waves). Grid: (B*NH, S/64).
// ---------------------------------------------------------------------------
__global__ void flash_attn(const bf16* __restrict__ q, const bf16* __restrict__ k,
                           const bf16* __restrict__ vt, bf16* __restrict__ ao) {
  __shared__ alignas(32) bf16 plds[4 * 16 * 32];  // per-wave P staging
  const int lane = threadIdx.x & 31;
  const int wave = threadIdx.x >> 5;
  const int bh    = blockIdx.x;                 // b*16 + head
  const int qrow0 = (blockIdx.y * 4 + wave) * 16;
  const int lm    = lane & 15;
  const int kb8   = (lane >> 4) * 8;
  const int kb16  = (lane >> 4) * 16;
  bf16* pl = plds + wave * 512;

  // Q^T as two 32x16 B-frags (d=0..31, d=32..63): lane = query row, loop-invariant
  const bf16* qp = q + ((size_t)bh * Sc + qrow0 + lm) * HDc;
  const v16bf bq0 = *(const v16bf*)(qp + kb16);
  const v16bf bq1 = *(const v16bf*)(qp + 32 + kb16);

  const bf16* kp = k  + (size_t)bh * Sc * HDc;
  const bf16* vp = vt + (size_t)bh * HDc * Sc;

  float m = -1e30f, l = 0.f;   // per-lane running stats for query row (lane&15)
  v8f o[4] = {};

  for (int j0 = 0; j0 < Sc; j0 += 32) {
    // A-frags: two 16-key tiles of K (rows = keys), split over d halves
    const bf16* kr0 = kp + (size_t)(j0 + lm) * HDc;
    const bf16* kr1 = kp + (size_t)(j0 + 16 + lm) * HDc;
    union U { v16bf v; v8bf h[2]; };
    U a00, a01, a10, a11;
    a00.h[0] = *(const v8bf*)(kr0 + kb16);       a00.h[1] = *(const v8bf*)(kr0 + kb16 + 8);
    a01.h[0] = *(const v8bf*)(kr0 + 32 + kb16);  a01.h[1] = *(const v8bf*)(kr0 + 40 + kb16);
    a10.h[0] = *(const v8bf*)(kr1 + kb16);       a10.h[1] = *(const v8bf*)(kr1 + kb16 + 8);
    a11.h[0] = *(const v8bf*)(kr1 + 32 + kb16);  a11.h[1] = *(const v8bf*)(kr1 + 40 + kb16);
    if (j0 + 64 < Sc) __builtin_prefetch(kp + (size_t)(j0 + 64 + lm) * HDc, 0, 1);

    v8f z = {};
    v8f s0 = WMMA_BF16(a00.v, bq0, z);  s0 = WMMA_BF16(a01.v, bq1, s0);
    v8f s1 = WMMA_BF16(a10.v, bq0, z);  s1 = WMMA_BF16(a11.v, bq1, s1);

    // in-lane max over this lane's 16 score values, one cross-half combine
    float cmax = -1e30f;
#pragma unroll
    for (int r = 0; r < 8; r++) {
      s0[r] *= 0.125f;  s1[r] *= 0.125f;        // 1/sqrt(HD)
      cmax = fmaxf(cmax, fmaxf(s0[r], s1[r]));
    }
    cmax = fmaxf(cmax, __shfl_xor(cmax, 16, 32));

    const float mn = fmaxf(m, cmax);
    const float al = __expf(m - mn);
    m = mn;

    float csum = 0.f;
    v8bf pb0, pb1;
#pragma unroll
    for (int r = 0; r < 8; r++) {
      const float e0 = __expf(s0[r] - mn);
      const float e1 = __expf(s1[r] - mn);
      csum += e0 + e1;
      pb0[r] = (bf16)e0;
      pb1[r] = (bf16)e1;
    }
    csum += __shfl_xor(csum, 16, 32);
    l = l * al + csum;

    // broadcast alpha per output row (uniform-index shuffles -> readlane)
    float aV[8];
#pragma unroll
    for (int r = 0; r < 8; r++) {
      const float lo = __shfl(al, r, 32);
      const float hi = __shfl(al, r + 8, 32);
      aV[r] = (lane & 16) ? hi : lo;
    }
#pragma unroll
    for (int t = 0; t < 4; t++)
#pragma unroll
      for (int r = 0; r < 8; r++) o[t][r] *= aV[r];

    // P^T (D-layout) packs contiguously into A-layout rows: 2 b128 DS stores
    *(v8bf*)(pl + lm * 32 + kb8)      = pb0;   // keys  0..15 of chunk
    *(v8bf*)(pl + lm * 32 + 16 + kb8) = pb1;   // keys 16..31 of chunk
    asm volatile("s_wait_dscnt 0" ::: "memory");
    U pa;
    pa.h[0] = *(const v8bf*)(pl + lm * 32 + kb16);
    pa.h[1] = *(const v8bf*)(pl + lm * 32 + kb16 + 8);

    // O[16x64] += P(16x32) @ V(32x64): 4 WMMAs, V^T rows are contiguous B-frags
#pragma unroll
    for (int t = 0; t < 4; t++) {
      const v16bf bvf = *(const v16bf*)(vp + (size_t)(t * 16 + lm) * Sc + j0 + kb16);
      o[t] = WMMA_BF16(pa.v, bvf, o[t]);
    }
  }

  // finalize: broadcast 1/l per row, write merged-head bf16 [B,S,H]
  const float inv = 1.f / l;
  float iV[8];
#pragma unroll
  for (int r = 0; r < 8; r++) {
    const float lo = __shfl(inv, r, 32);
    const float hi = __shfl(inv, r + 8, 32);
    iV[r] = (lane & 16) ? hi : lo;
  }
  const int bb = bh >> 4, nh = bh & 15;
#pragma unroll
  for (int r = 0; r < 8; r++) {
    const int s = qrow0 + r + ((lane >> 4) << 3);
#pragma unroll
    for (int t = 0; t < 4; t++)
      ao[(((size_t)(bb * Sc + s)) << 10) + nh * 64 + t * 16 + lm] = (bf16)(o[t][r] * iV[r]);
  }
}

// ---------------------------------------------------------------------------
// Launcher
// ---------------------------------------------------------------------------
extern "C" void kernel_launch(void* const* d_in, const int* in_sizes, int n_in,
                              void* d_out, int out_size, void* d_ws, size_t ws_size,
                              hipStream_t stream) {
  const float* x   = (const float*)d_in[0];
  const float* wq  = (const float*)d_in[1];  const float* bq  = (const float*)d_in[2];
  const float* wk  = (const float*)d_in[3];  const float* bk  = (const float*)d_in[4];
  const float* wv  = (const float*)d_in[5];  const float* bvp = (const float*)d_in[6];
  const float* wo  = (const float*)d_in[7];  const float* bo  = (const float*)d_in[8];
  const float* w1  = (const float*)d_in[9];  const float* b1  = (const float*)d_in[10];
  const float* w2  = (const float*)d_in[11]; const float* b2  = (const float*)d_in[12];
  const float* g1  = (const float*)d_in[13]; const float* be1 = (const float*)d_in[14];
  const float* g2  = (const float*)d_in[15]; const float* be2 = (const float*)d_in[16];
  float* out = (float*)d_out;

  char* ws = (char*)d_ws;
  size_t off = 0;
  auto alloc = [&](size_t bytes) -> void* {
    void* p = ws + off;
    off += (bytes + 255) & ~(size_t)255;
    return p;
  };
  bf16* h1   = (bf16*)alloc((size_t)Mrows * Hc * 2);
  bf16* wqT  = (bf16*)alloc((size_t)Hc * Hc * 2);
  bf16* wkT  = (bf16*)alloc((size_t)Hc * Hc * 2);
  bf16* wvT  = (bf16*)alloc((size_t)Hc * Hc * 2);
  bf16* woT  = (bf16*)alloc((size_t)Hc * Hc * 2);
  bf16* w1T  = (bf16*)alloc((size_t)Hc * Fc * 2);
  bf16* w2T  = (bf16*)alloc((size_t)Fc * Hc * 2);
  bf16* qh   = (bf16*)alloc((size_t)Mrows * Hc * 2);   // [B,NH,S,HD]
  bf16* kh   = (bf16*)alloc((size_t)Mrows * Hc * 2);   // [B,NH,S,HD]
  bf16* vth  = (bf16*)alloc((size_t)Mrows * Hc * 2);   // [B,NH,HD,S]
  bf16* ao   = (bf16*)alloc((size_t)Mrows * Hc * 2);   // [B,S,H]
  float* x2  = (float*)alloc((size_t)Mrows * Hc * 4);
  bf16* h2   = (bf16*)alloc((size_t)Mrows * Hc * 2);
  bf16* f1   = (bf16*)alloc((size_t)Mrows * Fc * 2);

  // weights -> transposed bf16
  transpose_convert<<<dim3(Hc / 32, Hc / 32), 256, 0, stream>>>(wq, wqT, Hc, Hc);
  transpose_convert<<<dim3(Hc / 32, Hc / 32), 256, 0, stream>>>(wk, wkT, Hc, Hc);
  transpose_convert<<<dim3(Hc / 32, Hc / 32), 256, 0, stream>>>(wv, wvT, Hc, Hc);
  transpose_convert<<<dim3(Hc / 32, Hc / 32), 256, 0, stream>>>(wo, woT, Hc, Hc);
  transpose_convert<<<dim3(Fc / 32, Hc / 32), 256, 0, stream>>>(w1, w1T, Hc, Fc);
  transpose_convert<<<dim3(Hc / 32, Fc / 32), 256, 0, stream>>>(w2, w2T, Fc, Hc);

  // LN1
  ln_kernel<<<Mrows, 256, 0, stream>>>(x, g1, be1, h1);

  // QKV projections (fused bias + head-split / V-transpose epilogues)
  gemm_wmma<MODE_HEAD><<<dim3(Mrows / 64, Hc / 128), 128, 0, stream>>>(
      h1, wqT, bq, nullptr, qh, Mrows, Hc, Hc);
  gemm_wmma<MODE_HEAD><<<dim3(Mrows / 64, Hc / 128), 128, 0, stream>>>(
      h1, wkT, bk, nullptr, kh, Mrows, Hc, Hc);
  gemm_wmma<MODE_VT><<<dim3(Mrows / 64, Hc / 128), 128, 0, stream>>>(
      h1, wvT, bvp, nullptr, vth, Mrows, Hc, Hc);

  // flash attention
  flash_attn<<<dim3(Bc * NHc, Sc / 64), 128, 0, stream>>>(qh, kh, vth, ao);

  // out projection + residual -> x2
  gemm_wmma<MODE_RESF32><<<dim3(Mrows / 64, Hc / 128), 128, 0, stream>>>(
      ao, woT, bo, x, x2, Mrows, Hc, Hc);

  // LN2
  ln_kernel<<<Mrows, 256, 0, stream>>>(x2, g2, be2, h2);

  // FFN1 with exact GELU
  gemm_wmma<MODE_GELU><<<dim3(Mrows / 64, Fc / 128), 128, 0, stream>>>(
      h2, w1T, b1, nullptr, f1, Mrows, Fc, Hc);

  // FFN2 + residual -> out
  gemm_wmma<MODE_RESF32><<<dim3(Mrows / 64, Hc / 128), 128, 0, stream>>>(
      f1, w2T, b2, x2, out, Mrows, Hc, Fc);
}
